// MPLayer_75118978007050
// MI455X (gfx1250) — compile-verified, matrix operand order checked by reading, not run on hardware
//
#include <hip/hip_runtime.h>

// ---- problem constants (match reference) ----
static constexpr int NB   = 128;  // batch
static constexpr int NN   = 64;   // nodes
static constexpr int NF   = 64;   // node features
static constexpr int KE0  = 129;  // edge MLP fan-in (2F+1)
static constexpr int KE0P = 160;  // padded to multiple of 32
static constexpr int E1   = 256;  // edge hidden
static constexpr int E3   = 96;   // edge out
static constexpr int KN0  = 160;  // node MLP fan-in (96 + 64), already 32-aligned
static constexpr int NM1  = 256;  // node hidden
static constexpr int NOUT = 64;   // node out
static constexpr int ROWS = NB * NN;  // 8192 node rows
static constexpr float ALPHA = 0.2f;

// LDS row strides (halfs), padded for bank-conflict-free ds_load_b128
static constexpr int AST = 168;   // A rows (160 logical)
static constexpr int HST = 264;   // H rows (256 logical)
static constexpr int WST = 40;    // Wt rows (32 logical + 8 pad) == TDM pad 4 DW / 16 DW

typedef __attribute__((ext_vector_type(16))) __bf16 v16bf;
typedef __attribute__((ext_vector_type(8)))  __bf16 v8bf;
typedef __attribute__((ext_vector_type(8)))  float  v8f;
typedef __attribute__((ext_vector_type(4)))  unsigned int u32x4;
typedef __attribute__((ext_vector_type(4)))  int i32x4;
typedef __attribute__((ext_vector_type(8)))  int i32x8;

#if defined(__HIP_DEVICE_COMPILE__) && __has_builtin(__builtin_amdgcn_tensor_load_to_lds)
#define HAVE_TDM 1
#else
#define HAVE_TDM 0
#endif

__device__ __forceinline__ float lrelu(float v) { return v > 0.f ? v : ALPHA * v; }

// Build a 16-half WMMA operand from two contiguous 8-half (16B) LDS chunks.
// Per ISA 7.12.2 (16-bit 16x32 operand): element e <-> K = 16*(e/8) + 8*h + (e%8).
__device__ __forceinline__ v16bf load_frag16(const __bf16* p0, const __bf16* p1) {
    v8bf lo = *(const v8bf*)p0;
    v8bf hi = *(const v8bf*)p1;
    v16bf r;
#pragma unroll
    for (int e = 0; e < 8; ++e) { r[e] = lo[e]; r[8 + e] = hi[e]; }
    return r;
}

#if HAVE_TDM
// TDM: DMA one (32 x NW) bf16 K-slice of W^T (row stride KPAD halfs) into LDS,
// inserting 4 DWORDs of pad every 16 DWORDs -> LDS row stride 40 halfs.
// D# layout per CDNA5 ISA 8.3/8.4.
template <int NW, int KPAD>
__device__ __forceinline__ void tdm_stage(const __bf16* __restrict__ WTg, int k0,
                                          __bf16 (*dst)[WST]) {
    const unsigned long long ga =
        (unsigned long long)(uintptr_t)WTg + (unsigned long long)k0 * 2ull;
    const unsigned int la = (unsigned int)(uintptr_t)&dst[0][0];
    u32x4 g0;
    g0[0] = 1u;                                               // count=1, user mode
    g0[1] = la;                                               // lds_addr (bytes)
    g0[2] = (unsigned int)ga;                                 // global_addr lo
    g0[3] = ((unsigned int)(ga >> 32) & 0x01FFFFFFu) | 0x80000000u; // hi | type=2
    i32x8 g1;
    g1[0] = (int)((1u << 16)      // data_size = 1 -> 2 bytes
                | (1u << 20)      // pad_enable
                | (3u << 22)      // pad_interval: every 16 DWORDs
                | (3u << 25));    // pad_amount: 4 DWORDs
    g1[1] = (int)(((unsigned)KPAD & 0xFFFFu) << 16);          // tensor_dim0 lo16
    g1[2] = (int)(((unsigned)NW & 0xFFFFu) << 16);            // dim0 hi=0 | tensor_dim1 lo16
    g1[3] = (int)(32u << 16);                                 // dim1 hi=0 | tile_dim0=32
    g1[4] = NW;                                               // tile_dim1=NW, tile_dim2=0
    g1[5] = KPAD;                                             // tensor_dim0_stride lo32
    g1[6] = 0;
    g1[7] = 0;
    i32x4 z4; z4[0] = z4[1] = z4[2] = z4[3] = 0;
#if __clang_major__ >= 23
    i32x8 z8;
#pragma unroll
    for (int e = 0; e < 8; ++e) z8[e] = 0;
    __builtin_amdgcn_tensor_load_to_lds(g0, g1, z4, z4, z8, 0);
#else
    __builtin_amdgcn_tensor_load_to_lds(g0, g1, z4, z4, 0);
#endif
}
#endif // HAVE_TDM

// One MLP layer as 64xNW GEMM over K = 32*KSTEPS, bf16 WMMA, f32 accum.
// Weight K-slices DMA'd by TDM from pre-transposed bf16 W^T (row stride KPAD
// halfs) into a double-buffered LDS tile; wave 0 drives TDM + tensorcnt, one
// workgroup barrier per K-step (the next iteration's barrier orders buffer
// reuse). Epilogue MODE: 0 = bias+lrelu -> bf16 Hout; 1 = bias+lrelu ->
// column-reduce into aggOut (ds_add_f32); 2 = bias (linear) -> f32 global Og.
template <int NW, int KSTEPS, int KPAD, int TILES, int MODE>
__device__ __forceinline__ void wmma_layer(
    const __bf16* __restrict__ Ab, int Astr,
    const __bf16* __restrict__ WTg, const float* __restrict__ bias,
    __bf16 (* __restrict__ Hout)[HST], float* __restrict__ aggOut,
    float* __restrict__ Og, int rowBase,
    __bf16 (* __restrict__ WtBuf)[E1][WST],
    int tid, int lane, int wave)
{
    const int mt = wave & 3;
    const int ng = wave >> 2;
    const int hh = lane >> 4;   // half-wave (K-interleave select)
    const int ln = lane & 15;   // row (A) / column (B,D) within tile
    const int m  = mt * 16 + ln;

    v8f acc[TILES];
#pragma unroll
    for (int t = 0; t < TILES; ++t)
#pragma unroll
        for (int e = 0; e < 8; ++e) acc[t][e] = 0.f;

    __syncthreads();   // prev layer: H writes visible, WtBuf readers done
#if HAVE_TDM
    if (wave == 0) tdm_stage<NW, KPAD>(WTg, 0, WtBuf[0]);
#endif

    for (int ks = 0; ks < KSTEPS; ++ks) {
#if HAVE_TDM
        if (wave == 0) __builtin_amdgcn_s_wait_tensorcnt(0);
        __syncthreads();                       // slice ks visible; prev buffer free
        if (wave == 0 && ks + 1 < KSTEPS)
            tdm_stage<NW, KPAD>(WTg, (ks + 1) * 32, WtBuf[(ks + 1) & 1]);
#else
        __syncthreads();                       // readers of this buffer done
        for (int idx = tid; idx < NW * 32; idx += 256) {
            const int c = idx >> 5, r = idx & 31;
            WtBuf[ks & 1][c][r] = WTg[c * KPAD + ks * 32 + r];
        }
        __syncthreads();
#endif
        const __bf16 (*Wt)[WST] = WtBuf[ks & 1];
        const __bf16* arow = Ab + m * Astr + ks * 32;
        v16bf af = load_frag16(arow + 8 * hh, arow + 16 + 8 * hh);
        v16bf bfr[TILES];
#pragma unroll
        for (int t = 0; t < TILES; ++t) {
            const __bf16* wrow = &Wt[(ng * TILES + t) * 16 + ln][0];
            bfr[t] = load_frag16(wrow + 8 * hh, wrow + 16 + 8 * hh);
        }
#pragma unroll
        for (int t = 0; t < TILES; ++t)
            acc[t] = __builtin_amdgcn_wmma_f32_16x16x32_bf16(
                false, af, false, bfr[t], (short)0, acc[t], false, false);
    }

#pragma unroll
    for (int t = 0; t < TILES; ++t) {
        const int n  = (ng * TILES + t) * 16 + ln;
        const float bv = bias[n];
        if (MODE == 0) {
#pragma unroll
            for (int r = 0; r < 8; ++r) {      // D layout: M = r + 8*hh
                float v = lrelu(acc[t][r] + bv);
                Hout[mt * 16 + 8 * hh + r][n] = (__bf16)v;
            }
        } else if (MODE == 1) {
            float s = 0.f;
#pragma unroll
            for (int r = 0; r < 8; ++r) s += lrelu(acc[t][r] + bv);
            atomicAdd(&aggOut[n], s);          // ds_add_f32
        } else {
#pragma unroll
            for (int r = 0; r < 8; ++r) {
                const int row = rowBase + mt * 16 + 8 * hh + r;
                Og[(size_t)row * NW + n] = acc[t][r] + bv;
            }
        }
    }
}

// One-shot: convert all MLP weights to bf16, transposed (W^T[n][k]); edge
// layer-0 K zero-padded to KE0P. Outputs live in d_ws (L2-resident).
extern "C" __global__ __launch_bounds__(256) void mp_convert_weights(
    const float* __restrict__ w0, const float* __restrict__ w1,
    const float* __restrict__ w2, const float* __restrict__ nw0,
    const float* __restrict__ nw1,
    __bf16* __restrict__ w0t, __bf16* __restrict__ w1t, __bf16* __restrict__ w2t,
    __bf16* __restrict__ w3t, __bf16* __restrict__ w4t)
{
    const int i = blockIdx.x * 256 + threadIdx.x;
    constexpr int S0 = E1 * KE0P;      // 40960
    constexpr int S1 = E1 * E1;        // 65536
    constexpr int S2 = E3 * E1;        // 24576
    constexpr int S3 = NM1 * KN0;      // 40960
    constexpr int S4 = NOUT * NM1;     // 16384
    if (i < S0) {
        const int c = i / KE0P, k = i % KE0P;
        w0t[i] = (k < KE0) ? (__bf16)w0[k * E1 + c] : (__bf16)0.f;
    } else if (i < S0 + S1) {
        const int j = i - S0;
        w1t[j] = (__bf16)w1[(j % E1) * E1 + (j / E1)];
    } else if (i < S0 + S1 + S2) {
        const int j = i - S0 - S1;
        w2t[j] = (__bf16)w2[(j % E1) * E3 + (j / E1)];
    } else if (i < S0 + S1 + S2 + S3) {
        const int j = i - S0 - S1 - S2;
        w3t[j] = (__bf16)nw0[(j % KN0) * NM1 + (j / KN0)];
    } else if (i < S0 + S1 + S2 + S3 + S4) {
        const int j = i - S0 - S1 - S2 - S3;
        w4t[j] = (__bf16)nw1[(j % NM1) * NOUT + (j / NM1)];
    }
}

// Kernel 1: per-(b,i) edge MLP; emits bf16 concat row [agg(96) | x_i(64)].
extern "C" __global__ __launch_bounds__(256) void mp_edge_kernel(
    const float* __restrict__ x,
    const __bf16* __restrict__ w0t, const float* __restrict__ feB0,
    const __bf16* __restrict__ w1t, const float* __restrict__ feB1,
    const __bf16* __restrict__ w2t, const float* __restrict__ feB2,
    __bf16* __restrict__ Yg)
{
    __shared__ __bf16 A_lds[NN][AST];       // [xi | xj | d | 0-pad] per row j
    __shared__ __bf16 H1[NN][HST];
    __shared__ __bf16 H2[NN][HST];
    __shared__ __bf16 WtBuf[2][E1][WST];    // TDM double buffer
    __shared__ float  aggL[E3];
    __shared__ float  xif[NF];

    const int b    = blockIdx.x / NN;
    const int i    = blockIdx.x % NN;
    const int tid  = threadIdx.x;
    const int lane = tid & 31;
    const int wave = tid >> 5;

    // ---- phase 0: load x[b] (bf16 xj columns), capture xi in f32 ----
    {
        const float* xb = x + (size_t)b * NN * NF;
        const int j = tid >> 2;
        const int c = (tid & 3) * 16;
#pragma unroll
        for (int k = 0; k < 16; ++k) {
            float v = xb[j * NF + c + k];
            A_lds[j][NF + c + k] = (__bf16)v;
            if (j == i) xif[c + k] = v;
        }
        if (tid < E3) aggL[tid] = 0.f;
    }
    __syncthreads();
    // xi broadcast into cols [0,64), zero pad cols [129,160)
    {
        const int j = tid >> 2;
        const int c = (tid & 3) * 16;
#pragma unroll
        for (int k = 0; k < 16; ++k) A_lds[j][c + k] = (__bf16)xif[c + k];
        if (tid < NN) {
            for (int k = KE0; k < KE0P; ++k) A_lds[tid][k] = (__bf16)0.f;
        }
    }
    __syncthreads();
    // per-row distance d_j = ||x_j - x_i + 1e-12||
    if (tid < NN) {
        float s = 0.f;
#pragma unroll
        for (int k = 0; k < NF; ++k) {
            float d = (float)A_lds[tid][NF + k] - xif[k] + 1e-12f;
            s += d * d;
        }
        A_lds[tid][2 * NF] = (__bf16)__builtin_sqrtf(s);
    }
    // (entry barrier inside wmma_layer covers the writes above)

    // ---- edge MLP: 3 WMMA GEMM layers, TDM-fed weights ----
    wmma_layer<E1, KE0P / 32, KE0P, 8, 0>(&A_lds[0][0], AST, w0t, feB0, H1, nullptr, nullptr, 0, WtBuf, tid, lane, wave);
    wmma_layer<E1, E1   / 32, E1  , 8, 0>(&H1[0][0],   HST, w1t, feB1, H2, nullptr, nullptr, 0, WtBuf, tid, lane, wave);
    wmma_layer<E3, E1   / 32, E1  , 3, 1>(&H2[0][0],   HST, w2t, feB2, nullptr, aggL, nullptr, 0, WtBuf, tid, lane, wave);

    __syncthreads();
    // emit Y row = [agg | xi] as bf16 for the node-MLP GEMM kernel
    {
        const size_t row = (size_t)b * NN + i;
        if (tid < E3)       Yg[row * KN0 + tid] = (__bf16)aggL[tid];
        else if (tid < KN0) Yg[row * KN0 + tid] = (__bf16)xif[tid - E3];
    }
}

// Kernel 2: node MLP as WMMA GEMM over 64-row blocks of Y.
extern "C" __global__ __launch_bounds__(256) void mp_node_kernel(
    const __bf16* __restrict__ Yg,
    const __bf16* __restrict__ w3t, const float* __restrict__ fnB0,
    const __bf16* __restrict__ w4t, const float* __restrict__ fnB1,
    float* __restrict__ out)
{
    __shared__ __bf16 A_lds[NN][AST];
    __shared__ __bf16 H1[NN][HST];
    __shared__ __bf16 WtBuf[2][E1][WST];

    const int tid  = threadIdx.x;
    const int lane = tid & 31;
    const int wave = tid >> 5;
    const int rowBase = blockIdx.x * NN;

    // cooperative load: 64 rows x 160 bf16 (16B chunks) into padded A_lds
    for (int idx = tid; idx < NN * (KN0 / 8); idx += 256) {
        const int r = idx / (KN0 / 8), c8 = idx % (KN0 / 8);
        *(v8bf*)&A_lds[r][c8 * 8] =
            *(const v8bf*)&Yg[((size_t)(rowBase + r)) * KN0 + c8 * 8];
    }
    // (entry barrier inside wmma_layer covers the writes above)

    wmma_layer<NM1,  KN0 / 32, KN0, 8, 0>(&A_lds[0][0], AST, w3t, fnB0, H1, nullptr, nullptr, 0, WtBuf, tid, lane, wave);
    wmma_layer<NOUT, NM1 / 32, NM1, 2, 2>(&H1[0][0],   HST, w4t, fnB1, nullptr, nullptr, out, rowBase, WtBuf, tid, lane, wave);
}

extern "C" void kernel_launch(void* const* d_in, const int* in_sizes, int n_in,
                              void* d_out, int out_size, void* d_ws, size_t ws_size,
                              hipStream_t stream) {
    (void)in_sizes; (void)n_in; (void)ws_size; (void)out_size;
    const float* x    = (const float*)d_in[0];
    const float* feW0 = (const float*)d_in[1];
    const float* feB0 = (const float*)d_in[2];
    const float* feW1 = (const float*)d_in[3];
    const float* feB1 = (const float*)d_in[4];
    const float* feW2 = (const float*)d_in[5];
    const float* feB2 = (const float*)d_in[6];
    const float* fnW0 = (const float*)d_in[7];
    const float* fnB0 = (const float*)d_in[8];
    const float* fnW1 = (const float*)d_in[9];
    const float* fnB1 = (const float*)d_in[10];
    float* out = (float*)d_out;

    // carve bf16 workspace: transposed weights + concat rows Y
    __bf16* w0t = (__bf16*)d_ws;                        // [256][160]
    __bf16* w1t = w0t + E1 * KE0P;                      // [256][256]
    __bf16* w2t = w1t + E1 * E1;                        // [ 96][256]
    __bf16* w3t = w2t + E3 * E1;                        // [256][160]
    __bf16* w4t = w3t + NM1 * KN0;                      // [ 64][256]
    __bf16* Yg  = w4t + NOUT * NM1;                     // [8192][160]

    constexpr int CONV_ELEMS =
        E1 * KE0P + E1 * E1 + E3 * E1 + NM1 * KN0 + NOUT * NM1;  // 188416
    hipLaunchKernelGGL(mp_convert_weights, dim3((CONV_ELEMS + 255) / 256), dim3(256),
                       0, stream, feW0, feW1, feW2, fnW0, fnW1,
                       w0t, w1t, w2t, w3t, w4t);

    hipLaunchKernelGGL(mp_edge_kernel, dim3(ROWS), dim3(256), 0, stream,
                       x, w0t, feB0, w1t, feB1, w2t, feB2, Yg);

    hipLaunchKernelGGL(mp_node_kernel, dim3(ROWS / NN), dim3(256), 0, stream,
                       Yg, w3t, fnB0, w4t, fnB1, out);
}
